// SNN_MLP_50852412784703
// MI455X (gfx1250) — matrix operand order repaired and checked by reading, not compiled, not run on hardware
//
#include <hip/hip_runtime.h>

typedef __attribute__((ext_vector_type(16))) _Float16 v16h;
typedef __attribute__((ext_vector_type(8)))  float    v8f;

#define SNN_B    32768
#define SNN_IN   784
#define SNN_H    128
#define SNN_OUT  10
#define SNN_THR  1.0f
#define SNN_BETA 0.9f
#define KC1      25     // ceil(784/32); last chunk half-valid (784 = 24*32 + 16)
#define KC2      4      // 128/32

// WMMA 16-bit fragment layouts (CDNA5 ISA 7.12.2, wave32):
//  A (16x32, MxK): lane -> M = lane%16 ; element e(=2v+h) -> K = (e<8) ? 8*(lane/16)+e
//                                                                     : 16+8*(lane/16)+(e-8)
//  B (32x16, KxN): lane -> N = lane%16 ; element e -> K as for A above (same split order)
//  C/D f32 (16x16): lane -> N = lane%16 ; vgpr v -> M = v + 8*(lane/16)

// ---- pack W1 (f32 [128,784]) into f16 B-fragments: w1p[kc][ct][lane][e] ----
__global__ __launch_bounds__(256) void snn_pack_w1(const float* __restrict__ W1,
                                                   _Float16* __restrict__ w1p) {
  int idx = blockIdx.x * 256 + threadIdx.x;          // 25*8*32*16 = 102400
  if (idx >= KC1 * 8 * 32 * 16) return;
  int e    = idx & 15;
  int lane = (idx >> 4) & 31;
  int ct   = (idx >> 9) & 7;
  int kc   = idx >> 12;
  int n = ct * 16 + (lane & 15);                     // H column (B = W1^T -> B[k][n] = W1[n][k])
  int k = kc * 32 + ((e < 8) ? (lane >> 4) * 8 + e : 16 + (lane >> 4) * 8 + (e - 8));
  w1p[idx] = (k < SNN_IN) ? (_Float16)W1[n * SNN_IN + k] : (_Float16)0.f;
}

// ---- pack W2 (f32 [10,128]) into f16 B-fragments, N padded 10->16 ----
__global__ __launch_bounds__(256) void snn_pack_w2(const float* __restrict__ W2,
                                                   _Float16* __restrict__ w2p) {
  int idx = blockIdx.x * 256 + threadIdx.x;          // 4*32*16 = 2048
  if (idx >= KC2 * 32 * 16) return;
  int e    = idx & 15;
  int lane = (idx >> 4) & 31;
  int kc   = idx >> 9;
  int n = lane & 15;
  int k = kc * 32 + ((e < 8) ? (lane >> 4) * 8 + e : 16 + (lane >> 4) * 8 + (e - 8));
  w2p[idx] = (n < SNN_OUT) ? (_Float16)W2[n * SNN_H + k] : (_Float16)0.f;
}

struct __align__(16) H8 { _Float16 h[8]; };

// ---- fused: GEMM1 (WMMA f16) + 25-step LIF recurrence + GEMM2 (WMMA f16) ----
__global__ __launch_bounds__(128) void snn_fused(
    const float* __restrict__ x,  const float* __restrict__ b1,
    const float* __restrict__ b2, const _Float16* __restrict__ w1p,
    const _Float16* __restrict__ w2p, const int* __restrict__ nsteps_p,
    float* __restrict__ out) {
  __shared__ _Float16 sT[4][SNN_H * 16];             // per-wave spk1 transpose buffer (4 KB each)

  const int lane = threadIdx.x & 31;
  const int wave = threadIdx.x >> 5;
  const int m    = lane & 15;                        // A-row / C-col index
  const int kg   = lane >> 4;                        // lane-half
  const int bm   = kg * 8;                           // C/D row base for this lane-half
  const int row0 = (blockIdx.x * 4 + wave) * 16;     // 16 batch rows per wave

  // ================= GEMM1: cur1 = x @ W1^T (+ b1) =================
  v8f acc[8] = {};                                   // 8 N-tiles of 16 -> H=128
  const float* xr = x + (size_t)(row0 + m) * SNN_IN;

  for (int kc = 0; kc < KC1; ++kc) {
    const int kb = kc * 32 + kg * 8;
    if (kc + 1 < KC1) __builtin_prefetch(xr + kb + 32, 0, 0);
    float4 f0 = *(const float4*)(xr + kb);
    float4 f1 = *(const float4*)(xr + kb + 4);
    v16h a;
    a[0] = (_Float16)f0.x; a[1] = (_Float16)f0.y; a[2] = (_Float16)f0.z; a[3] = (_Float16)f0.w;
    a[4] = (_Float16)f1.x; a[5] = (_Float16)f1.y; a[6] = (_Float16)f1.z; a[7] = (_Float16)f1.w;
    if (kc < KC1 - 1) {
      float4 f2 = *(const float4*)(xr + kb + 16);
      float4 f3 = *(const float4*)(xr + kb + 20);
      a[8]  = (_Float16)f2.x; a[9]  = (_Float16)f2.y; a[10] = (_Float16)f2.z; a[11] = (_Float16)f2.w;
      a[12] = (_Float16)f3.x; a[13] = (_Float16)f3.y; a[14] = (_Float16)f3.z; a[15] = (_Float16)f3.w;
    } else {                                         // K 784..799 zero pad
      #pragma unroll
      for (int e = 8; e < 16; ++e) a[e] = (_Float16)0.f;
    }
    const _Float16* wp = w1p + (size_t)kc * 4096 + (size_t)lane * 16;
    #pragma unroll
    for (int ct = 0; ct < 8; ++ct) {
      v16h bf = *(const v16h*)(wp + ct * 512);       // contiguous 32B per lane, L2-resident
      acc[ct] = __builtin_amdgcn_wmma_f32_16x16x32_f16(
          false, a, false, bf, (short)0, acc[ct], false, false);
    }
  }

  #pragma unroll
  for (int ct = 0; ct < 8; ++ct) {
    float bias = b1[ct * 16 + m];
    #pragma unroll
    for (int j = 0; j < 8; ++j) acc[ct][j] += bias;  // acc == cur1, constant across steps
  }

  // resident W2 B-fragments + padded b2
  v16h w2f[KC2];
  #pragma unroll
  for (int kc = 0; kc < KC2; ++kc)
    w2f[kc] = *(const v16h*)(w2p + ((size_t)kc * 32 + lane) * 16);
  const float b2v = (m < SNN_OUT) ? b2[m] : 0.f;

  // ================= LIF recurrence =================
  v8f mem1[8] = {};
  v8f mem2 = {};
  _Float16* s = sT[wave];
  const int nsteps = *nsteps_p;

  for (int t = 0; t < nsteps; ++t) {
    // layer 1: reset(old mem) -> decay+integrate -> spike; spk1 -> LDS transposed [h][m]
    #pragma unroll
    for (int ct = 0; ct < 8; ++ct) {
      H8 sp;
      #pragma unroll
      for (int j = 0; j < 8; ++j) {
        float mm = mem1[ct][j];
        float r  = (mm > SNN_THR) ? SNN_THR : 0.f;
        mm = SNN_BETA * mm + acc[ct][j] - r;
        mem1[ct][j] = mm;
        sp.h[j] = (mm > SNN_THR) ? (_Float16)1.f : (_Float16)0.f;
      }
      int hcol = ct * 16 + m;                        // H index of this lane
      *(H8*)(s + hcol * 16 + bm) = sp;               // ds_store_b128, 16B aligned
    }
    asm volatile("s_wait_dscnt 0" ::: "memory");     // wave-local LDS fence (same-wave, no barrier)

    // layer 2 GEMM: cur2 = spk1 @ W2^T ; rebuild A-fragments from LDS
    v8f c2 = {};
    #pragma unroll
    for (int kc = 0; kc < KC2; ++kc) {
      v16h a2;
      #pragma unroll
      for (int e = 0; e < 8; ++e)
        a2[e] = s[(kc * 32 + kg * 8 + e) * 16 + m];
      #pragma unroll
      for (int e = 0; e < 8; ++e)
        a2[8 + e] = s[(kc * 32 + 16 + kg * 8 + e) * 16 + m];
      c2 = __builtin_amdgcn_wmma_f32_16x16x32_f16(
          false, a2, false, w2f[kc], (short)0, c2, false, false);
    }

    // layer 2 LIF + output spikes [t, row, o]
    #pragma unroll
    for (int j = 0; j < 8; ++j) {
      float cur2 = c2[j] + b2v;
      float mm = mem2[j];
      float r  = (mm > SNN_THR) ? SNN_THR : 0.f;
      mm = SNN_BETA * mm + cur2 - r;
      mem2[j] = mm;
      float sp2 = (mm > SNN_THR) ? 1.f : 0.f;
      if (m < SNN_OUT) {
        size_t row = (size_t)(row0 + bm + j);
        out[(size_t)t * SNN_B * SNN_OUT + row * SNN_OUT + m] = sp2;
      }
    }
  }
}

extern "C" void kernel_launch(void* const* d_in, const int* in_sizes, int n_in,
                              void* d_out, int out_size, void* d_ws, size_t ws_size,
                              hipStream_t stream) {
  const float* x      = (const float*)d_in[0];   // [32768,1,28,28]
  const float* W1     = (const float*)d_in[1];   // [128,784]
  const float* b1     = (const float*)d_in[2];   // [128]
  const float* W2     = (const float*)d_in[3];   // [10,128]
  const float* b2     = (const float*)d_in[4];   // [10]
  const int*   nsteps = (const int*)d_in[5];     // 25
  float* out = (float*)d_out;                    // [25,32768,10]

  _Float16* w1p = (_Float16*)d_ws;                         // 102400 halves = 200 KB
  _Float16* w2p = (_Float16*)((char*)d_ws + 204800);       // 2048 halves = 4 KB

  snn_pack_w1<<<400, 256, 0, stream>>>(W1, w1p);
  snn_pack_w2<<<8,   256, 0, stream>>>(W2, w2p);
  // 32768 rows / (16 rows/wave * 4 waves/block) = 512 blocks
  snn_fused<<<512, 128, 0, stream>>>(x, b1, b2, w1p, w2p, nsteps, out);
}
